// AttentionWithSink_24292335027125
// MI455X (gfx1250) — matrix operand order repaired and checked by reading, not compiled, not run on hardware
//
#include <hip/hip_runtime.h>
#include <hip/hip_bf16.h>

// ---------------- problem constants ----------------
constexpr int B  = 2;
constexpr int S  = 2048;
constexpr int D  = 1024;
constexpr int H  = 16;
constexpr int HD = 64;          // head dim
constexpr int NS = 8;           // sink tokens
constexpr int T  = S + NS;      // 2056 total tokens
constexpr int TP = 2080;        // padded tokens: 65*32 (also 130*16)
constexpr int MT32_P = TP / 32; // 65 row tiles (32 rows) for projections
constexpr int NQ   = D / 64;    // 16 column "quads" (one head each)
constexpr int QT32 = TP / 32;   // 65 query tiles of 32
constexpr int KT   = TP / 32;   // 65 key tiles of 32

typedef __attribute__((ext_vector_type(16))) _Float16 v16h;
typedef __attribute__((ext_vector_type(8)))  _Float16 v8h;
typedef __attribute__((ext_vector_type(8)))  float    v8f;

union AF { v16h v; v8h h[2]; };

__device__ __forceinline__ v8h ld8(const _Float16* p) { return *(const v8h*)p; }

__device__ __forceinline__ v8f wmma_f16(const AF& a, const AF& b, v8f c) {
    return __builtin_amdgcn_wmma_f32_16x16x32_f16(false, a.v, false, b.v,
                                                  (short)0, c, false, false);
}

// ---------------- kernel 1: build padded f16 xs = [sinks ; x ; 0-pad] ----------------
__global__ void build_xs_kernel(const float* __restrict__ x,
                                const float* __restrict__ sink,
                                _Float16* __restrict__ xs)
{
    size_t i = (size_t)blockIdx.x * blockDim.x + threadIdx.x;
    if (i >= (size_t)B * TP * D) return;
    int d = i % D;
    int t = (i / D) % TP;
    int b = i / ((size_t)TP * D);
    float v;
    if (t < NS)        v = sink[t * D + d];
    else if (t < T)    v = x[((size_t)b * S + (t - NS)) * D + d];
    else               v = 0.0f;
    xs[i] = (_Float16)v;
}

// ---------------- kernel 2: f32 -> f16 weight conversion ----------------
__global__ void cvt_f16_kernel(const float* __restrict__ in, _Float16* __restrict__ out, int n)
{
    int i = blockIdx.x * blockDim.x + threadIdx.x;
    if (i < n) out[i] = (_Float16)in[i];
}

// ---------------- kernel 3: projection GEMM  Y = xs @ W^T + b  (32x64 per wave) ----------
// Software-pipelined k-loop; optionally stores V transposed [b,h,hd,Tp].
__global__ __launch_bounds__(128) void proj_wmma_kernel(
    const _Float16* __restrict__ xs, const _Float16* __restrict__ w,
    const float* __restrict__ bias, _Float16* __restrict__ out,
    float scale, int transposedV)
{
    int wave = blockIdx.x * (blockDim.x >> 5) + (threadIdx.x >> 5);
    int nq = wave % NQ;
    int mt = (wave / NQ) % MT32_P;
    int b  = wave / (NQ * MT32_P);
    int l   = threadIdx.x & 31;
    int m16 = l & 15;
    int hi  = l >> 4;

    const _Float16* arow[2];
    const _Float16* wrow[4];
#pragma unroll
    for (int mi = 0; mi < 2; ++mi)
        arow[mi] = xs + ((size_t)b * TP + mt * 32 + mi * 16 + m16) * D;
#pragma unroll
    for (int j = 0; j < 4; ++j)
        wrow[j] = w + (size_t)(nq * 64 + j * 16 + m16) * D;

    float bj[4];
#pragma unroll
    for (int j = 0; j < 4; ++j) bj[j] = bias[nq * 64 + j * 16 + m16];

    v8f acc[2][4] = {};
    AF aC[2], bC[4];
#pragma unroll
    for (int mi = 0; mi < 2; ++mi) {
        aC[mi].h[0] = ld8(arow[mi] + 8 * hi);
        aC[mi].h[1] = ld8(arow[mi] + 16 + 8 * hi);
    }
#pragma unroll
    for (int j = 0; j < 4; ++j) {
        bC[j].h[0] = ld8(wrow[j] + 8 * hi);
        bC[j].h[1] = ld8(wrow[j] + 16 + 8 * hi);
    }

#pragma unroll 2
    for (int k = 32; k < D; k += 32) {
        AF aN[2], bN[4];                         // prefetch k while computing k-32
#pragma unroll
        for (int mi = 0; mi < 2; ++mi) {
            aN[mi].h[0] = ld8(arow[mi] + k + 8 * hi);
            aN[mi].h[1] = ld8(arow[mi] + k + 16 + 8 * hi);
        }
#pragma unroll
        for (int j = 0; j < 4; ++j) {
            bN[j].h[0] = ld8(wrow[j] + k + 8 * hi);
            bN[j].h[1] = ld8(wrow[j] + k + 16 + 8 * hi);
        }
#pragma unroll
        for (int mi = 0; mi < 2; ++mi)
#pragma unroll
            for (int j = 0; j < 4; ++j)
                acc[mi][j] = wmma_f16(aC[mi], bC[j], acc[mi][j]);
#pragma unroll
        for (int mi = 0; mi < 2; ++mi) aC[mi] = aN[mi];
#pragma unroll
        for (int j = 0; j < 4; ++j) bC[j] = bN[j];
    }
#pragma unroll
    for (int mi = 0; mi < 2; ++mi)
#pragma unroll
        for (int j = 0; j < 4; ++j)
            acc[mi][j] = wmma_f16(aC[mi], bC[j], acc[mi][j]);

#pragma unroll
    for (int mi = 0; mi < 2; ++mi) {
#pragma unroll
        for (int j = 0; j < 4; ++j) {
#pragma unroll
            for (int r = 0; r < 8; ++r) {
                int row = mt * 32 + mi * 16 + r + 8 * hi;  // C: M=r+8*hi, N=lane&15
                int hd  = j * 16 + m16;
                float vv = (acc[mi][j][r] + bj[j]) * scale;
                size_t idx = transposedV
                    ? ((size_t)(b * H + nq) * HD + hd) * TP + row
                    : ((size_t)(b * H + nq) * TP + row) * HD + hd;
                out[idx] = (_Float16)vv;
            }
        }
    }
}

// ---------------- kernel 4: flash attention with sinks (32 queries / wave) ----------------
__global__ __launch_bounds__(128) void attn_wmma_kernel(
    const _Float16* __restrict__ q, const _Float16* __restrict__ k,
    const _Float16* __restrict__ vT, _Float16* __restrict__ o)
{
    __shared__ __align__(16) _Float16 pbuf[4][32 * 32];  // per-wave 32x32 p tile
    int wv   = threadIdx.x >> 5;
    int wave = blockIdx.x * (blockDim.x >> 5) + wv;
    int qt = wave % QT32;
    int h  = (wave / QT32) % H;
    int b  = wave / (QT32 * H);
    int l   = threadIdx.x & 31;
    int m16 = l & 15;
    int hi  = l >> 4;

    size_t bh = (size_t)(b * H + h);

    AF aq[2][2];                         // two 16-query subtiles x (K 0..31, 32..63)
#pragma unroll
    for (int qa = 0; qa < 2; ++qa) {
        const _Float16* qrow = q + (bh * TP + qt * 32 + qa * 16 + m16) * HD;
        aq[qa][0].h[0] = ld8(qrow + 8 * hi);
        aq[qa][0].h[1] = ld8(qrow + 16 + 8 * hi);
        aq[qa][1].h[0] = ld8(qrow + 32 + 8 * hi);
        aq[qa][1].h[1] = ld8(qrow + 48 + 8 * hi);
    }

    float mrow[2][8], lrow[2][8];
#pragma unroll
    for (int qa = 0; qa < 2; ++qa)
#pragma unroll
        for (int r = 0; r < 8; ++r) { mrow[qa][r] = -1e30f; lrow[qa][r] = 0.0f; }
    v8f oacc[2][4] = {};
    _Float16* pb = pbuf[wv];

    auto loadK = [&](int kbase, AF bk[2][2]) {
#pragma unroll
        for (int cA = 0; cA < 2; ++cA) {
            const _Float16* krow = k + (bh * TP + kbase + cA * 16 + m16) * HD + 8 * hi;
            bk[cA][0].h[0] = ld8(krow);          // kd 0..31  chunk
            bk[cA][0].h[1] = ld8(krow + 16);
            bk[cA][1].h[0] = ld8(krow + 32);     // kd 32..63 chunk
            bk[cA][1].h[1] = ld8(krow + 48);
        }
    };

    AF bk[2][2];
    loadK(0, bk);

    for (int kt = 0; kt < KT; ++kt) {
        int kbase = kt * 32;

        // V fragments issued early; consumed after the softmax + LDS round trip
        AF bv[4];
#pragma unroll
        for (int c4 = 0; c4 < 4; ++c4) {
            const _Float16* vrow = vT + (bh * HD + c4 * 16 + m16) * TP + kbase + 8 * hi;
            bv[c4].h[0] = ld8(vrow);
            bv[c4].h[1] = ld8(vrow + 16);
        }

        // scores: 4 WMMAs (2 query subtiles x 2 key subtiles), K=64 via chain
        v8f sc[2][2];
#pragma unroll
        for (int qa = 0; qa < 2; ++qa)
#pragma unroll
            for (int cA = 0; cA < 2; ++cA) {
                v8f c = {};
                c = wmma_f16(aq[qa][0], bk[cA][0], c);
                c = wmma_f16(aq[qa][1], bk[cA][1], c);
                sc[qa][cA] = c;
            }

        // prefetch next key tile's fragments (clamped; last-iter result unused)
        AF bkn[2][2];
        int knext = kbase + 32;
        if (knext > TP - 32) knext = 0;
        loadK(knext, bkn);

        bool va = (kbase + m16) < T;             // mask padded keys
        bool vb = (kbase + 16 + m16) < T;

        float p0[2][8], p1[2][8], fs[2][8];
#pragma unroll
        for (int qa = 0; qa < 2; ++qa) {
#pragma unroll
            for (int r = 0; r < 8; ++r) {
                float s0 = va ? sc[qa][0][r] : -1e30f;
                float s1 = vb ? sc[qa][1][r] : -1e30f;
                float lm = fmaxf(s0, s1);
                lm = fmaxf(lm, __shfl_xor(lm, 1, 32));  // row lives in a 16-lane half
                lm = fmaxf(lm, __shfl_xor(lm, 2, 32));
                lm = fmaxf(lm, __shfl_xor(lm, 4, 32));
                lm = fmaxf(lm, __shfl_xor(lm, 8, 32));
                float mn = fmaxf(mrow[qa][r], lm);
                float f  = exp2f(mrow[qa][r] - mn);
                float e0 = va ? exp2f(s0 - mn) : 0.0f;
                float e1 = vb ? exp2f(s1 - mn) : 0.0f;
                float rs = e0 + e1;
                rs += __shfl_xor(rs, 1, 32);
                rs += __shfl_xor(rs, 2, 32);
                rs += __shfl_xor(rs, 4, 32);
                rs += __shfl_xor(rs, 8, 32);
                lrow[qa][r] = lrow[qa][r] * f + rs;
                mrow[qa][r] = mn;
                fs[qa][r] = f; p0[qa][r] = e0; p1[qa][r] = e1;
            }
#pragma unroll
            for (int c4 = 0; c4 < 4; ++c4)
#pragma unroll
                for (int r = 0; r < 8; ++r)
                    oacc[qa][c4][r] = oacc[qa][c4][r] * fs[qa][r];
        }

        // C-layout -> A-layout transpose through per-wave LDS; same-wave LDS ops are
        // in-order (ISA 7.3), so a wave-local DScnt wait + compiler barrier suffices.
#pragma unroll
        for (int qa = 0; qa < 2; ++qa)
#pragma unroll
            for (int r = 0; r < 8; ++r) {
                int row = qa * 16 + r + 8 * hi;
                pb[row * 32 + m16]      = (_Float16)p0[qa][r];
                pb[row * 32 + 16 + m16] = (_Float16)p1[qa][r];
            }
        asm volatile("s_wait_dscnt 0x0" ::: "memory");
        AF ap[2];
#pragma unroll
        for (int qa = 0; qa < 2; ++qa) {
            ap[qa].h[0] = ld8(pb + (qa * 16 + m16) * 32 + 8 * hi);
            ap[qa].h[1] = ld8(pb + (qa * 16 + m16) * 32 + 16 + 8 * hi);
        }

        // o += P(16x32) x V(32x16) per hd chunk: 8 WMMAs
#pragma unroll
        for (int qa = 0; qa < 2; ++qa)
#pragma unroll
            for (int c4 = 0; c4 < 4; ++c4)
                oacc[qa][c4] = wmma_f16(ap[qa], bv[c4], oacc[qa][c4]);

#pragma unroll
        for (int cA = 0; cA < 2; ++cA) {
            bk[cA][0] = bkn[cA][0];
            bk[cA][1] = bkn[cA][1];
        }
    }

#pragma unroll
    for (int qa = 0; qa < 2; ++qa) {
        float inv[8];
#pragma unroll
        for (int r = 0; r < 8; ++r) inv[r] = 1.0f / lrow[qa][r];
#pragma unroll
        for (int c4 = 0; c4 < 4; ++c4) {
#pragma unroll
            for (int r = 0; r < 8; ++r) {
                int qr = qt * 32 + qa * 16 + r + 8 * hi;
                if (qr >= NS && qr < T) {        // drop sink rows + padding
                    int srow = qr - NS;
                    int d = h * HD + c4 * 16 + m16;
                    o[((size_t)b * S + srow) * D + d] = (_Float16)(oacc[qa][c4][r] * inv[r]);
                }
            }
        }
    }
}

// ---------------- kernel 5: output projection  out = attn @ Wo^T + bo (f32 out) -------------
__global__ __launch_bounds__(128) void oproj_wmma_kernel(
    const _Float16* __restrict__ a_in, const _Float16* __restrict__ w,
    const float* __restrict__ bias, float* __restrict__ out)
{
    int wave = blockIdx.x * (blockDim.x >> 5) + (threadIdx.x >> 5);
    int nq = wave % NQ;
    int mt = wave / NQ;                  // 0..127 over B*S rows (32 each)
    int l   = threadIdx.x & 31;
    int m16 = l & 15;
    int hi  = l >> 4;

    const _Float16* arow[2];
    const _Float16* wrow[4];
#pragma unroll
    for (int mi = 0; mi < 2; ++mi)
        arow[mi] = a_in + ((size_t)mt * 32 + mi * 16 + m16) * D;
#pragma unroll
    for (int j = 0; j < 4; ++j)
        wrow[j] = w + (size_t)(nq * 64 + j * 16 + m16) * D;

    float bj[4];
#pragma unroll
    for (int j = 0; j < 4; ++j) bj[j] = bias[nq * 64 + j * 16 + m16];

    v8f acc[2][4] = {};
    AF aC[2], bC[4];
#pragma unroll
    for (int mi = 0; mi < 2; ++mi) {
        aC[mi].h[0] = ld8(arow[mi] + 8 * hi);
        aC[mi].h[1] = ld8(arow[mi] + 16 + 8 * hi);
    }
#pragma unroll
    for (int j = 0; j < 4; ++j) {
        bC[j].h[0] = ld8(wrow[j] + 8 * hi);
        bC[j].h[1] = ld8(wrow[j] + 16 + 8 * hi);
    }

#pragma unroll 2
    for (int k = 32; k < D; k += 32) {
        AF aN[2], bN[4];
#pragma unroll
        for (int mi = 0; mi < 2; ++mi) {
            aN[mi].h[0] = ld8(arow[mi] + k + 8 * hi);
            aN[mi].h[1] = ld8(arow[mi] + k + 16 + 8 * hi);
        }
#pragma unroll
        for (int j = 0; j < 4; ++j) {
            bN[j].h[0] = ld8(wrow[j] + k + 8 * hi);
            bN[j].h[1] = ld8(wrow[j] + k + 16 + 8 * hi);
        }
#pragma unroll
        for (int mi = 0; mi < 2; ++mi)
#pragma unroll
            for (int j = 0; j < 4; ++j)
                acc[mi][j] = wmma_f16(aC[mi], bC[j], acc[mi][j]);
#pragma unroll
        for (int mi = 0; mi < 2; ++mi) aC[mi] = aN[mi];
#pragma unroll
        for (int j = 0; j < 4; ++j) bC[j] = bN[j];
    }
#pragma unroll
    for (int mi = 0; mi < 2; ++mi)
#pragma unroll
        for (int j = 0; j < 4; ++j)
            acc[mi][j] = wmma_f16(aC[mi], bC[j], acc[mi][j]);

#pragma unroll
    for (int mi = 0; mi < 2; ++mi)
#pragma unroll
        for (int j = 0; j < 4; ++j)
#pragma unroll
            for (int r = 0; r < 8; ++r) {
                size_t row = (size_t)mt * 32 + mi * 16 + r + 8 * hi;
                int n = nq * 64 + j * 16 + m16;
                out[row * D + n] = acc[mi][j][r] + bj[j];
            }
}

// ---------------- host launch ----------------
extern "C" void kernel_launch(void* const* d_in, const int* in_sizes, int n_in,
                              void* d_out, int out_size, void* d_ws, size_t ws_size,
                              hipStream_t stream)
{
    (void)in_sizes; (void)n_in; (void)out_size; (void)ws_size;
    const float* x    = (const float*)d_in[0];
    const float* sink = (const float*)d_in[1];
    const float* wq = (const float*)d_in[2];
    const float* bq = (const float*)d_in[3];
    const float* wk = (const float*)d_in[4];
    const float* bk = (const float*)d_in[5];
    const float* wv = (const float*)d_in[6];
    const float* bv = (const float*)d_in[7];
    const float* wo = (const float*)d_in[8];
    const float* bo = (const float*)d_in[9];
    float* out = (float*)d_out;

    char* ws = (char*)d_ws;
    size_t off = 0;
    auto carve = [&](size_t bytes) -> void* {
        void* p = ws + off;
        off = (off + bytes + 255) & ~(size_t)255;
        return p;
    };
    _Float16* xs_h = (_Float16*)carve((size_t)B * TP * D * 2);
    _Float16* wq_h = (_Float16*)carve((size_t)D * D * 2);
    _Float16* wk_h = (_Float16*)carve((size_t)D * D * 2);
    _Float16* wv_h = (_Float16*)carve((size_t)D * D * 2);
    _Float16* wo_h = (_Float16*)carve((size_t)D * D * 2);
    _Float16* q_h  = (_Float16*)carve((size_t)B * H * TP * HD * 2);
    _Float16* k_h  = (_Float16*)carve((size_t)B * H * TP * HD * 2);
    _Float16* vT_h = (_Float16*)carve((size_t)B * H * HD * TP * 2);
    _Float16* at_h = (_Float16*)carve((size_t)B * S * D * 2);

    // 1) pack xs (sinks + x + zero pad) as f16
    {
        size_t n = (size_t)B * TP * D;
        build_xs_kernel<<<(unsigned)((n + 255) / 256), 256, 0, stream>>>(x, sink, xs_h);
    }
    // 2) convert weights
    {
        int n = D * D, g = (n + 255) / 256;
        cvt_f16_kernel<<<g, 256, 0, stream>>>(wq, wq_h, n);
        cvt_f16_kernel<<<g, 256, 0, stream>>>(wk, wk_h, n);
        cvt_f16_kernel<<<g, 256, 0, stream>>>(wv, wv_h, n);
        cvt_f16_kernel<<<g, 256, 0, stream>>>(wo, wo_h, n);
    }
    // 3) Q/K/V projections (q pre-scaled by 1/sqrt(HD) * log2(e) for exp2 softmax)
    {
        int waves = B * MT32_P * NQ;          // 2080
        int grid  = waves / 4;                // 4 waves (128 thr) per block
        const float qscale = 0.125f * 1.4426950408889634f;
        proj_wmma_kernel<<<grid, 128, 0, stream>>>(xs_h, wq_h, bq, q_h,  qscale, 0);
        proj_wmma_kernel<<<grid, 128, 0, stream>>>(xs_h, wk_h, bk, k_h,  1.0f,   0);
        proj_wmma_kernel<<<grid, 128, 0, stream>>>(xs_h, wv_h, bv, vT_h, 1.0f,   1);
    }
    // 4) attention (32 queries per wave)
    {
        int waves = B * H * QT32;             // 2080
        attn_wmma_kernel<<<waves / 4, 128, 0, stream>>>(q_h, k_h, vT_h, at_h);
    }
    // 5) output projection
    {
        int waves = (B * S / 32) * NQ;        // 2048
        oproj_wmma_kernel<<<waves / 4, 128, 0, stream>>>(at_h, wo_h, bo, out);
    }
}
